// SlidingWindowAttention_19224273617232
// MI455X (gfx1250) — matrix operand order repaired and checked by reading, not compiled, not run on hardware
//
#include <hip/hip_runtime.h>
#include <hip/hip_bf16.h>
#include <math.h>

// Problem constants (from reference)
constexpr int BB   = 2;
constexpr int SS   = 1024;
constexpr int DIN  = 512;
constexpr int DOUT = 512;
constexpr int HH   = 8;
constexpr int RR   = 32;
constexpr int WW   = 2 * RR + 1;   // 65
constexpr int DK   = DOUT / HH;    // 64
constexpr int NROW = BB * SS;      // 2048 GEMM rows

typedef __attribute__((ext_vector_type(16))) _Float16 v16h;
typedef __attribute__((ext_vector_type(8)))  _Float16 v8h;
typedef __attribute__((ext_vector_type(4)))  _Float16 v4h;
typedef __attribute__((ext_vector_type(8)))  float    v8f;

// ---------------------------------------------------------------------------
// f32 -> f16 pre-conversion (one pass; removes all cvt work from GEMM loops)
// ---------------------------------------------------------------------------
__global__ __launch_bounds__(256)
void cvt_f32_to_f16(const float* __restrict__ src, _Float16* __restrict__ dst, int n)
{
    const int i = ((int)blockIdx.x * 256 + (int)threadIdx.x) * 4;
    if (i + 3 < n) {
        const float4 f = *(const float4*)(src + i);
        v4h o = { (_Float16)f.x, (_Float16)f.y, (_Float16)f.z, (_Float16)f.w };
        *(v4h*)(dst + i) = o;
    }
}

// ---------------------------------------------------------------------------
// WMMA GEMM (f16 inputs, f32 accumulate):  Y[n,o] = sum_i A[n,i] * Wt[o,i]
// A: [M,512] f16 row-major, Wt: [512,512] f16 row-major.
// Each wave computes a 32x64 output block (2 M-tiles x 4 N-tiles -> 8 WMMAs
// per K-step of 32), register-reusing A across N and B across M.
// mode 0: dst[n*DOUT + o]  (plain row-major, final Wc projection)
// mode 1: dst[((b*H + h)*S + s)*DK + d]  with o = d*H + h  (split-heads)
// ---------------------------------------------------------------------------
__global__ __launch_bounds__(256)
void wmma_gemm_kernel(const _Float16* __restrict__ A,
                      const _Float16* __restrict__ Wt,
                      float* __restrict__ dst,
                      int mode)
{
    const int lane   = threadIdx.x & 31;
    const int waveId = (int)((blockIdx.x * blockDim.x + threadIdx.x) >> 5);
    const int tilesN = DOUT / 64;              // 8 column blocks
    const int tm     = waveId / tilesN;        // 32-row block  (0..63)
    const int tn     = waveId % tilesN;        // 64-col block  (0..7)
    const int hi     = lane >> 4;              // half-wave select
    const int l15    = lane & 15;

    // A-fragment rows for the two M-tiles; B-fragment cols for the 4 N-tiles.
    const _Float16* arow[2];
    arow[0] = A + (long)(tm * 32 + l15)      * DIN;
    arow[1] = A + (long)(tm * 32 + 16 + l15) * DIN;
    const _Float16* brow[4];
    #pragma unroll
    for (int ni = 0; ni < 4; ++ni)
        brow[ni] = Wt + (long)(tn * 64 + ni * 16 + l15) * DIN;

    v8f acc[2][4] = {};

    for (int kb = 0; kb < DIN; kb += 32) {
        // Prefetch next K-slice (lowers to global_prefetch_b8 on gfx1250).
        if (kb + 32 < DIN) {
            __builtin_prefetch(arow[0] + kb + 32, 0, 3);
            __builtin_prefetch(brow[0] + kb + 32, 0, 3);
        }
        // A fragments (16x32 f16): halves j -> K = kb + hi*16 + j (contiguous).
        v16h amat[2];
        #pragma unroll
        for (int mi = 0; mi < 2; ++mi) {
            const _Float16* ap = arow[mi] + kb + hi * 16;
            v8h lo = *(const v8h*)(ap);
            v8h hi8 = *(const v8h*)(ap + 8);
            amat[mi] = __builtin_shufflevector(lo, hi8,
                        0,1,2,3,4,5,6,7,8,9,10,11,12,13,14,15);
        }
        // B fragments (32x16 f16): halves j -> K = kb + hi*16 + j (contiguous).
        v16h bmat[4];
        #pragma unroll
        for (int ni = 0; ni < 4; ++ni)
            bmat[ni] = *(const v16h*)(brow[ni] + kb + hi * 16);

        #pragma unroll
        for (int mi = 0; mi < 2; ++mi)
            #pragma unroll
            for (int ni = 0; ni < 4; ++ni)
                acc[mi][ni] = __builtin_amdgcn_wmma_f32_16x16x32_f16(
                    false, amat[mi], false, bmat[ni],
                    (short)0, acc[mi][ni], false, false);
    }

    // D layout: VGPR v -> row = tileRow + v + hi*8, col = tileCol + l15.
    #pragma unroll
    for (int mi = 0; mi < 2; ++mi) {
        #pragma unroll
        for (int ni = 0; ni < 4; ++ni) {
            const int o = tn * 64 + ni * 16 + l15;
            #pragma unroll
            for (int v = 0; v < 8; ++v) {
                const int n   = tm * 32 + mi * 16 + v + hi * 8;
                const float y = acc[mi][ni][v];
                if (mode == 0) {
                    dst[(long)n * DOUT + o] = y;
                } else {
                    const int b = n / SS, s = n % SS;
                    const int h = o % HH, d = o / HH;   // o = d*H + h
                    dst[(((long)(b * HH + h) * SS) + s) * DK + d] = y;
                }
            }
        }
    }
}

// ---------------------------------------------------------------------------
// Sliding-window attention. One wave per (b,h,s) row, 8 waves / block.
// q,k,v in [b][h][s][d] layout (f32). Produces ctx[(b*S+s)*DOUT + d*H + h].
// ---------------------------------------------------------------------------
__global__ __launch_bounds__(256)
void swa_attn_kernel(const float* __restrict__ qh,
                     const float* __restrict__ kh,
                     const float* __restrict__ vh,
                     const int*   __restrict__ pm,   // padding_mask [B,S], nonzero = masked
                     const float* __restrict__ E,    // [H, W, DK]
                     float*       __restrict__ ctx)
{
    __shared__ float sh_q[8][DK];       // per-wave query vector
    __shared__ float sh_a[8][WW + 7];   // per-wave attention probs (padded)

    const int lane = threadIdx.x & 31;
    const int wave = threadIdx.x >> 5;
    const int row  = (int)blockIdx.x * 8 + wave;   // (b*H + h)*S + s
    const int s    = row % SS;
    const int bh   = row / SS;
    const int h    = bh % HH;
    const int b    = bh / HH;

    // Stage q into LDS (broadcast-friendly).
    const float* qrow = qh + (long)row * DK;
    sh_q[wave][lane]      = qrow[lane];
    sh_q[wave][lane + 32] = qrow[lane + 32];
    __syncthreads();

    // Each lane owns window slots: w = lane, lane+32, and (lane==0) w=64.
    const int nw = (lane == 0) ? 3 : 2;
    int   wl[3]   = { lane, lane + 32, 64 };
    float ev[3]   = { 0.f, 0.f, 0.f };
    bool  oorv[3] = { true, true, true };
    bool  pmv[3]  = { false, false, false };
    bool  allm = true;

    for (int t = 0; t < nw; ++t) {
        const int w = wl[t];
        const int p = s + w - RR;
        const bool oor = (p < 0) || (p >= SS);
        const bool pmw = (!oor) && (pm[b * SS + p] != 0);
        float acc = 0.f;
        if (!oor) {
            const float* kp = kh + (((long)(b * HH + h) * SS) + p) * DK;
            const float* ep = E + ((long)h * WW + w) * DK;
            #pragma unroll 8
            for (int d = 0; d < DK; ++d)
                acc += sh_q[wave][d] * (kp[d] + ep[d]);   // q·k_win + q·E (shared q)
        }
        ev[t]   = acc * 0.125f;                           // 1/sqrt(64)
        oorv[t] = oor;
        pmv[t]  = pmw;
        allm    = allm && (oor || pmw);
    }

    // m2 = pm_win & ~all(full_mask): if everything is masked, drop pm masking.
    const bool allMasked = (__all((int)allm) != 0);
    const float NEGINF = -__builtin_inff();
    for (int t = 0; t < nw; ++t) {
        const bool msk = oorv[t] || (pmv[t] && !allMasked);
        if (msk) ev[t] = NEGINF;
    }

    // Softmax over 65 window positions (wave32 shuffle reductions).
    float m = NEGINF;
    for (int t = 0; t < nw; ++t) m = fmaxf(m, ev[t]);
    #pragma unroll
    for (int off = 16; off > 0; off >>= 1)
        m = fmaxf(m, __shfl_xor(m, off, 32));

    float xv[3];
    float sum = 0.f;
    for (int t = 0; t < nw; ++t) { xv[t] = __expf(ev[t] - m); sum += xv[t]; }
    #pragma unroll
    for (int off = 16; off > 0; off >>= 1)
        sum += __shfl_xor(sum, off, 32);
    const float inv = 1.f / sum;

    for (int t = 0; t < nw; ++t) sh_a[wave][wl[t]] = xv[t] * inv;
    __syncthreads();

    // out[d] = sum_w attn[w] * v[p,d];  d = lane, lane+32 (coalesced v loads).
    const int d0 = lane, d1 = lane + 32;
    float a0 = 0.f, a1 = 0.f;
    for (int w = 0; w < WW; ++w) {
        const int p = s + w - RR;
        if (p >= 0 && p < SS) {
            const float aw = sh_a[wave][w];
            const float* vp = vh + (((long)(b * HH + h) * SS) + p) * DK;
            a0 += aw * vp[d0];
            a1 += aw * vp[d1];
        }
    }
    // ctx channel index o' = d*H + h (matches reference transpose/reshape)
    float* crow = ctx + ((long)(b * SS + s)) * DOUT;
    crow[d0 * HH + h] = a0;
    crow[d1 * HH + h] = a1;
}

// ---------------------------------------------------------------------------
extern "C" void kernel_launch(void* const* d_in, const int* in_sizes, int n_in,
                              void* d_out, int out_size, void* d_ws, size_t ws_size,
                              hipStream_t stream)
{
    const float* Q  = (const float*)d_in[0];
    const float* K  = (const float*)d_in[1];
    const float* V  = (const float*)d_in[2];
    const int*   PM = (const int*)  d_in[3];
    const float* Wq = (const float*)d_in[4];
    const float* Wk = (const float*)d_in[5];
    const float* Wv = (const float*)d_in[6];
    const float* Wc = (const float*)d_in[7];
    const float* E  = (const float*)d_in[8];
    float* out = (float*)d_out;

    // ---- workspace carving ------------------------------------------------
    const long ACT = (long)NROW * DIN;    // 1<<20 elements (activations/ctx)
    const long WEL = (long)DOUT * DIN;    // 1<<18 elements (weights)

    char* p = (char*)d_ws;
    float* qf   = (float*)p;  p += ACT * sizeof(float);   // q  [b,h,s,d] f32
    float* kf   = (float*)p;  p += ACT * sizeof(float);   // k
    float* vf   = (float*)p;  p += ACT * sizeof(float);   // v
    float* ctxf = (float*)p;  p += ACT * sizeof(float);   // ctx [B*S, DOUT] f32
    _Float16* Qh   = (_Float16*)p;  p += ACT * sizeof(_Float16);
    _Float16* Kh   = (_Float16*)p;  p += ACT * sizeof(_Float16);
    _Float16* Vh   = (_Float16*)p;  p += ACT * sizeof(_Float16);
    _Float16* Ch   = (_Float16*)p;  p += ACT * sizeof(_Float16);
    _Float16* Wqh  = (_Float16*)p;  p += WEL * sizeof(_Float16);
    _Float16* Wkh  = (_Float16*)p;  p += WEL * sizeof(_Float16);
    _Float16* Wvh  = (_Float16*)p;  p += WEL * sizeof(_Float16);
    _Float16* Wch  = (_Float16*)p;  p += WEL * sizeof(_Float16);

    // ---- 1) f32 -> f16 pre-conversion ------------------------------------
    const int actBlocks = (int)(ACT / 1024);   // 4 elems/thread, 256 threads
    const int wBlocks   = (int)(WEL / 1024);
    cvt_f32_to_f16<<<actBlocks, 256, 0, stream>>>(Q,  Qh,  (int)ACT);
    cvt_f32_to_f16<<<actBlocks, 256, 0, stream>>>(K,  Kh,  (int)ACT);
    cvt_f32_to_f16<<<actBlocks, 256, 0, stream>>>(V,  Vh,  (int)ACT);
    cvt_f32_to_f16<<<wBlocks,   256, 0, stream>>>(Wq, Wqh, (int)WEL);
    cvt_f32_to_f16<<<wBlocks,   256, 0, stream>>>(Wk, Wkh, (int)WEL);
    cvt_f32_to_f16<<<wBlocks,   256, 0, stream>>>(Wv, Wvh, (int)WEL);
    cvt_f32_to_f16<<<wBlocks,   256, 0, stream>>>(Wc, Wch, (int)WEL);

    // ---- 2) projections (WMMA) -------------------------------------------
    // (NROW/32) * (DOUT/64) = 64*8 = 512 waves -> 64 blocks of 8 waves.
    const int gemmBlocks = (NROW / 32) * (DOUT / 64) / 8;
    wmma_gemm_kernel<<<gemmBlocks, 256, 0, stream>>>(Qh, Wqh, qf, 1);
    wmma_gemm_kernel<<<gemmBlocks, 256, 0, stream>>>(Kh, Wkh, kf, 1);
    wmma_gemm_kernel<<<gemmBlocks, 256, 0, stream>>>(Vh, Wvh, vf, 1);

    // ---- 3) sliding-window attention -------------------------------------
    const int attnBlocks = (BB * HH * SS) / 8;  // 2048
    swa_attn_kernel<<<attnBlocks, 256, 0, stream>>>(qf, kf, vf, PM, E, ctxf);

    // ---- 4) output projection (WMMA) -------------------------------------
    cvt_f32_to_f16<<<actBlocks, 256, 0, stream>>>(ctxf, Ch, (int)ACT);
    wmma_gemm_kernel<<<gemmBlocks, 256, 0, stream>>>(Ch, Wch, out, 0);
}
